// VisionTransformer_10350871183578
// MI455X (gfx1250) — matrix operand order repaired and checked by reading, not compiled, not run on hardware
//
#include <hip/hip_runtime.h>
#include <hip/hip_bf16.h>
#include <math.h>

typedef __attribute__((ext_vector_type(16))) _Float16 v16h;
typedef __attribute__((ext_vector_type(8)))  float    v8f;

// ---- model dims ----
#define Bz    32
#define Cc_   3
#define IMG_  224
#define Pp    16
#define Dd    768
#define Hh    12
#define Ll    12
#define NCc   1000
#define Gg    14
#define Ss    197
#define SP    224            /* S padded to a multiple of 32 (K-dim padding) */
#define HDd   64
#define FFf   3072
#define NPATCH (Gg*Gg)       /* 196 */
#define MTOK  (Bz*Ss)        /* 6304 */

// ---- GEMM tiling ----
#define BM 128
#define BN 128
#define BK 32
#define LDS_STRIDE 40        /* 32 + 8 halves pad: 80B rows -> 16B aligned, 20-bank stride */

union HalfVec { uint4 u[2]; v16h h; };

// Async copy of 16 bytes global -> LDS (CDNA5 GLOBAL_LOAD_ASYNC_TO_LDS_B128,
// tracked by ASYNCcnt). lds_off = LDS byte address (low 32 bits of generic ptr).
__device__ __forceinline__ void async_copy_b128(unsigned lds_off, const void* gaddr)
{
    asm volatile("global_load_async_to_lds_b128 %0, %1, off"
                 :: "v"(lds_off), "v"(gaddr) : "memory");
}
__device__ __forceinline__ void wait_async0()
{
    asm volatile("s_wait_asynccnt 0x0" ::: "memory");
}
__device__ __forceinline__ unsigned lds_addr(const void* p)
{
    return (unsigned)(unsigned long long)p;   // LDS aperture: addr[31:0] = LDS offset
}

// ======================= async-staged WMMA GEMM =======================
// Requires: N % BN == 0, K % BK == 0, lda/ldb multiples of 8, A/Bt buffers
// padded past the last tile row. C = act(scale * A@B + bias + res).
__global__ __launch_bounds__(256)
void k_gemm_async(const _Float16* __restrict__ A, int lda, long long strideA,
                  const _Float16* __restrict__ Bt, int ldb, long long strideB,
                  float* Cf, _Float16* Ch, int ldc, long long strideC,
                  const float* __restrict__ bias, const float* res,
                  int M, int N, int K, float scale, int act)
{
    __shared__ __align__(16) _Float16 sA [2][BM][LDS_STRIDE];
    __shared__ __align__(16) _Float16 sBT[2][BN][LDS_STRIDE];

    const int bz = blockIdx.z;
    A  += (long long)bz * strideA;
    Bt += (long long)bz * strideB;
    const long long cOff = (long long)bz * strideC;

    const int tid  = threadIdx.x;
    const int lane = tid & 31;
    const int wave = tid >> 5;
    const int wm   = wave & 3;        // 4 waves along M (32 rows each)
    const int wn   = wave >> 2;       // 2 waves along N (64 cols each)
    const int hi   = lane >> 4;
    const int lm   = lane & 15;

    const int m0 = blockIdx.y * BM;
    const int n0 = blockIdx.x * BN;

    // staging: 2 b128 per thread per tile (128 rows x 32 cols each)
    const int r0 = tid >> 2;
    const int r1 = r0 + 64;
    const int c0 = (tid & 3) * 8;
    const _Float16* aP0 = A  + (long long)(m0 + r0) * lda + c0;
    const _Float16* aP1 = A  + (long long)(m0 + r1) * lda + c0;
    const _Float16* bP0 = Bt + (long long)(n0 + r0) * ldb + c0;
    const _Float16* bP1 = Bt + (long long)(n0 + r1) * ldb + c0;

    v8f acc[2][4] = {};

    // prologue: stage tile 0 into buffer 0
    async_copy_b128(lds_addr(&sA [0][r0][c0]), aP0);
    async_copy_b128(lds_addr(&sA [0][r1][c0]), aP1);
    async_copy_b128(lds_addr(&sBT[0][r0][c0]), bP0);
    async_copy_b128(lds_addr(&sBT[0][r1][c0]), bP1);
    wait_async0();
    __syncthreads();

    int cur = 0;
    for (int k0 = 0; k0 < K; k0 += BK) {
        const int kn = k0 + BK;
        if (kn < K) {   // kick async staging of next tile into the other buffer
            const int nb = cur ^ 1;
            async_copy_b128(lds_addr(&sA [nb][r0][c0]), aP0 + kn);
            async_copy_b128(lds_addr(&sA [nb][r1][c0]), aP1 + kn);
            async_copy_b128(lds_addr(&sBT[nb][r0][c0]), bP0 + kn);
            async_copy_b128(lds_addr(&sBT[nb][r1][c0]), bP1 + kn);
        }

        // ---- fragments from current buffer: contiguous 16B -> ds_load_b128 ----
        HalfVec bf[4];
        #pragma unroll
        for (int ni = 0; ni < 4; ++ni) {
            const _Float16* brow = &sBT[cur][wn * 64 + ni * 16 + lm][0];
            bf[ni].u[0] = *(const uint4*)(brow + (hi << 4));
            bf[ni].u[1] = *(const uint4*)(brow + (hi << 4) + 8);
        }
        #pragma unroll
        for (int mi = 0; mi < 2; ++mi) {
            const _Float16* arow = &sA[cur][wm * 32 + mi * 16 + lm][0];
            HalfVec af;
            af.u[0] = *(const uint4*)(arow + (hi << 4));
            af.u[1] = *(const uint4*)(arow + (hi << 4) + 8);
            #pragma unroll
            for (int ni = 0; ni < 4; ++ni) {
                acc[mi][ni] = __builtin_amdgcn_wmma_f32_16x16x32_f16(
                    false, af.h, false, bf[ni].h, (short)0, acc[mi][ni], false, false);
            }
        }

        wait_async0();          // own async stores to LDS done
        __syncthreads();        // everyone's done -> next buffer valid
        cur ^= 1;
    }

    // ---- epilogue: C/D layout — lane col n = lm, row m = v + 8*hi per 16x16 tile ----
    #pragma unroll
    for (int mi = 0; mi < 2; ++mi) {
        #pragma unroll
        for (int ni = 0; ni < 4; ++ni) {
            const int n = n0 + wn * 64 + ni * 16 + lm;
            const float bv = bias ? bias[n] : 0.0f;
            #pragma unroll
            for (int v = 0; v < 8; ++v) {
                const int m = m0 + wm * 32 + mi * 16 + v + (hi << 3);
                if (m >= M) continue;
                const long long o = cOff + (long long)m * ldc + n;
                float val = acc[mi][ni][v] * scale + bv;
                if (res) val += res[o];
                if (act == 1) val = 0.5f * val * (1.0f + erff(val * 0.70710678f));
                if (Ch) Ch[o] = (_Float16)val;
                else    Cf[o] = val;
            }
        }
    }
}

// ======================= generic (edge-N) WMMA GEMM =======================
// Register double-buffered; runtime N guards. Used for N % BN != 0.
__global__ __launch_bounds__(256)
void k_gemm_edge(const _Float16* __restrict__ A, int lda, long long strideA,
                 const _Float16* __restrict__ Bt, int ldb, long long strideB,
                 float* Cf, _Float16* Ch, int ldc, long long strideC,
                 const float* __restrict__ bias, const float* res,
                 int M, int N, int K, float scale, int act)
{
    __shared__ __align__(16) _Float16 sA [BM][LDS_STRIDE];
    __shared__ __align__(16) _Float16 sBT[BN][LDS_STRIDE];

    const int bz = blockIdx.z;
    A  += (long long)bz * strideA;
    Bt += (long long)bz * strideB;
    const long long cOff = (long long)bz * strideC;

    const int tid  = threadIdx.x;
    const int lane = tid & 31;
    const int wave = tid >> 5;
    const int wm   = wave & 3;
    const int wn   = wave >> 2;
    const int hi   = lane >> 4;
    const int lm   = lane & 15;

    const int m0 = blockIdx.y * BM;
    const int n0 = blockIdx.x * BN;
    const bool nFull = (n0 + BN) <= N;

    const int r0 = tid >> 2;
    const int r1 = r0 + 64;
    const int c0 = (tid & 3) * 8;

    uint4 ra0, ra1, rb0, rb1;
    _Float16 rbs[16];

    auto loadTiles = [&](int k0) {
        ra0 = *(const uint4*)(A + (long long)(m0 + r0) * lda + k0 + c0);
        ra1 = *(const uint4*)(A + (long long)(m0 + r1) * lda + k0 + c0);
        if (nFull) {
            rb0 = *(const uint4*)(Bt + (long long)(n0 + r0) * ldb + k0 + c0);
            rb1 = *(const uint4*)(Bt + (long long)(n0 + r1) * ldb + k0 + c0);
        } else {
            #pragma unroll
            for (int i = 0; i < 16; ++i) {
                const int idx = i * 256 + tid;
                const int r = idx >> 5, c = idx & 31;
                rbs[i] = (n0 + r < N) ? Bt[(long long)(n0 + r) * ldb + k0 + c]
                                      : (_Float16)0.0f;
            }
        }
    };
    auto storeTiles = [&]() {
        *(uint4*)&sA[r0][c0] = ra0;
        *(uint4*)&sA[r1][c0] = ra1;
        if (nFull) {
            *(uint4*)&sBT[r0][c0] = rb0;
            *(uint4*)&sBT[r1][c0] = rb1;
        } else {
            #pragma unroll
            for (int i = 0; i < 16; ++i) {
                const int idx = i * 256 + tid;
                sBT[idx >> 5][idx & 31] = rbs[i];
            }
        }
    };

    v8f acc[2][4] = {};

    loadTiles(0);
    for (int k0 = 0; k0 < K; k0 += BK) {
        storeTiles();
        __syncthreads();

        const int kn = k0 + BK;
        if (kn < K) loadTiles(kn);

        HalfVec bf[4];
        #pragma unroll
        for (int ni = 0; ni < 4; ++ni) {
            const _Float16* brow = &sBT[wn * 64 + ni * 16 + lm][0];
            bf[ni].u[0] = *(const uint4*)(brow + (hi << 4));
            bf[ni].u[1] = *(const uint4*)(brow + (hi << 4) + 8);
        }
        #pragma unroll
        for (int mi = 0; mi < 2; ++mi) {
            const _Float16* arow = &sA[wm * 32 + mi * 16 + lm][0];
            HalfVec af;
            af.u[0] = *(const uint4*)(arow + (hi << 4));
            af.u[1] = *(const uint4*)(arow + (hi << 4) + 8);
            #pragma unroll
            for (int ni = 0; ni < 4; ++ni) {
                acc[mi][ni] = __builtin_amdgcn_wmma_f32_16x16x32_f16(
                    false, af.h, false, bf[ni].h, (short)0, acc[mi][ni], false, false);
            }
        }
        __syncthreads();
    }

    #pragma unroll
    for (int mi = 0; mi < 2; ++mi) {
        #pragma unroll
        for (int ni = 0; ni < 4; ++ni) {
            const int n = n0 + wn * 64 + ni * 16 + lm;
            if (n >= N) continue;
            const float bv = bias ? bias[n] : 0.0f;
            #pragma unroll
            for (int v = 0; v < 8; ++v) {
                const int m = m0 + wm * 32 + mi * 16 + v + (hi << 3);
                if (m >= M) continue;
                const long long o = cOff + (long long)m * ldc + n;
                float val = acc[mi][ni][v] * scale + bv;
                if (res) val += res[o];
                if (act == 1) val = 0.5f * val * (1.0f + erff(val * 0.70710678f));
                if (Ch) Ch[o] = (_Float16)val;
                else    Cf[o] = val;
            }
        }
    }
}

// ---- LayerNorm over D=768, one block per token row, f16 output ----
__global__ __launch_bounds__(256)
void k_layernorm(const float* __restrict__ x, const float* __restrict__ s,
                 const float* __restrict__ b, _Float16* __restrict__ out)
{
    __shared__ float red[256];
    const long long row = blockIdx.x;
    const float* xr = x + row * Dd;
    float sum = 0.f;
    for (int i = threadIdx.x; i < Dd; i += 256) sum += xr[i];
    red[threadIdx.x] = sum; __syncthreads();
    for (int st = 128; st > 0; st >>= 1) {
        if (threadIdx.x < st) red[threadIdx.x] += red[threadIdx.x + st];
        __syncthreads();
    }
    const float mean = red[0] / Dd; __syncthreads();
    float vs = 0.f;
    for (int i = threadIdx.x; i < Dd; i += 256) { float d = xr[i] - mean; vs += d * d; }
    red[threadIdx.x] = vs; __syncthreads();
    for (int st = 128; st > 0; st >>= 1) {
        if (threadIdx.x < st) red[threadIdx.x] += red[threadIdx.x + st];
        __syncthreads();
    }
    const float rstd = rsqrtf(red[0] / Dd + 1e-5f);
    for (int i = threadIdx.x; i < Dd; i += 256)
        out[row * Dd + i] = (_Float16)((xr[i] - mean) * rstd * s[i] + b[i]);
}

// ---- row softmax: reads f32 [rows x Ss], writes f16 [rows x SP] (zero padded) ----
__global__ __launch_bounds__(256)
void k_softmax(const float* __restrict__ sc, _Float16* __restrict__ p)
{
    __shared__ float red[256];
    const long long row = blockIdx.x;
    const float* r = sc + row * Ss;
    float mx = -1e30f;
    for (int i = threadIdx.x; i < Ss; i += 256) mx = fmaxf(mx, r[i]);
    red[threadIdx.x] = mx; __syncthreads();
    for (int st = 128; st > 0; st >>= 1) {
        if (threadIdx.x < st) red[threadIdx.x] = fmaxf(red[threadIdx.x], red[threadIdx.x + st]);
        __syncthreads();
    }
    mx = red[0]; __syncthreads();
    float sum = 0.f;
    for (int i = threadIdx.x; i < Ss; i += 256) sum += __expf(r[i] - mx);
    red[threadIdx.x] = sum; __syncthreads();
    for (int st = 128; st > 0; st >>= 1) {
        if (threadIdx.x < st) red[threadIdx.x] += red[threadIdx.x + st];
        __syncthreads();
    }
    const float inv = 1.0f / red[0];
    for (int i = threadIdx.x; i < SP; i += 256)
        p[row * SP + i] = (i < Ss) ? (_Float16)(__expf(r[i] - mx) * inv) : (_Float16)0.0f;
}

// ---- im2col: images (B,C,224,224) f32 -> patch matrix [B*196, 768] f16 ----
__global__ void k_im2col(const float* __restrict__ img, _Float16* __restrict__ out)
{
    const long long i = (long long)blockIdx.x * 256 + threadIdx.x;
    const long long total = (long long)Bz * NPATCH * (Cc_ * Pp * Pp);
    if (i >= total) return;
    const int col = (int)(i % (Cc_ * Pp * Pp));
    const long long bp = i / (Cc_ * Pp * Pp);
    const int b = (int)(bp / NPATCH), pidx = (int)(bp % NPATCH);
    const int gh = pidx / Gg, gw = pidx % Gg;
    const int c = col >> 8, pr = (col >> 4) & 15, q = col & 15;
    out[i] = (_Float16)img[(((long long)b * Cc_ + c) * IMG_ + gh * Pp + pr) * IMG_ + gw * Pp + q];
}

// ---- x[b,s,d] = (s==0 ? cls[d] : xp[b,s-1,d]) + PE(s,d) ----
__global__ void k_assemble_x(const float* __restrict__ xp, const float* __restrict__ cls,
                             float* __restrict__ x)
{
    const long long i = (long long)blockIdx.x * 256 + threadIdx.x;
    if (i >= (long long)Bz * Ss * Dd) return;
    const int d = (int)(i % Dd);
    const long long t = i / Dd;
    const int s = (int)(t % Ss), b = (int)(t / Ss);
    const int de = d & ~1;
    const float ang = (float)s * __expf(-(float)de * (9.210340372f / (float)Dd)); // 10000^(-de/D)
    const float pe = (d & 1) ? __cosf(ang) : __sinf(ang);
    const float v = (s == 0) ? cls[d] : xp[((long long)b * NPATCH + (s - 1)) * Dd + d];
    x[i] = v + pe;
}

// ---- conversions ----
__global__ void k_f32_to_f16(const float* __restrict__ src, _Float16* __restrict__ dst, long long n)
{
    const long long i = (long long)blockIdx.x * 256 + threadIdx.x;
    if (i < n) dst[i] = (_Float16)src[i];
}
__global__ void k_transpose_to_f16(const float* __restrict__ src, _Float16* __restrict__ dst,
                                   int R, int Cn)  // src [R,Cn] -> dst [Cn,R]
{
    const long long i = (long long)blockIdx.x * 256 + threadIdx.x;
    if (i >= (long long)R * Cn) return;
    const int c = (int)(i % Cn), r = (int)(i / Cn);
    dst[(long long)c * R + r] = (_Float16)src[i];
}

// ---- split QKV into attention layouts ----
__global__ void k_qkv_split(const _Float16* __restrict__ q, const _Float16* __restrict__ k,
                            const _Float16* __restrict__ v, _Float16* __restrict__ Qh,
                            _Float16* __restrict__ Kh, _Float16* __restrict__ VhT)
{
    const long long i = (long long)blockIdx.x * 256 + threadIdx.x;
    if (i >= (long long)Bz * Hh * HDd * SP) return;
    const int s = (int)(i % SP);
    long long t = i / SP;
    const int hd = (int)(t % HDd); t /= HDd;
    const int h  = (int)(t % Hh);
    const int b  = (int)(t / Hh);
    const long long bh = (long long)b * Hh + h;
    const long long src = ((long long)b * Ss + s) * Dd + h * HDd + hd;
    if (s < Ss) {
        const _Float16 qv = q[src];
        const _Float16 kv = k[src];
        Qh[(bh * Ss + s) * HDd + hd] = qv;
        Kh[(bh * Ss + s) * HDd + hd] = kv;
        VhT[(bh * HDd + hd) * SP + s] = v[src];
    } else {
        VhT[(bh * HDd + hd) * SP + s] = (_Float16)0.0f;
    }
}

__global__ void k_merge_heads(const _Float16* __restrict__ Oh, _Float16* __restrict__ Om)
{
    const long long i = (long long)blockIdx.x * 256 + threadIdx.x;
    if (i >= (long long)Bz * Ss * Dd) return;
    const int d = (int)(i % Dd);
    const long long t = i / Dd;
    const int s = (int)(t % Ss), b = (int)(t / Ss);
    const int h = d / HDd, hd = d % HDd;
    Om[i] = Oh[(((long long)b * Hh + h) * Ss + s) * HDd + hd];
}

__global__ void k_extract_cls(const float* __restrict__ x, _Float16* __restrict__ cls16)
{
    const int i = blockIdx.x * 256 + threadIdx.x;
    if (i >= Bz * Dd) return;
    const int d = i % Dd, b = i / Dd;
    cls16[i] = (_Float16)x[((long long)b * Ss) * Dd + d];
}

// ---- host-side helpers ----
static inline long long ceilDiv(long long a, long long b) { return (a + b - 1) / b; }

static void gemm(hipStream_t st,
                 const _Float16* A, int lda, long long sA,
                 const _Float16* Bt, int ldb, long long sB,
                 float* Cf, _Float16* Ch, int ldc, long long sC,
                 const float* bias, const float* res,
                 int M, int N, int K, float scale, int act, int batch)
{
    dim3 grid((unsigned)ceilDiv(N, BN), (unsigned)ceilDiv(M, BM), (unsigned)batch);
    if (N % BN == 0)
        k_gemm_async<<<grid, 256, 0, st>>>(A, lda, sA, Bt, ldb, sB, Cf, Ch, ldc, sC,
                                           bias, res, M, N, K, scale, act);
    else
        k_gemm_edge<<<grid, 256, 0, st>>>(A, lda, sA, Bt, ldb, sB, Cf, Ch, ldc, sC,
                                          bias, res, M, N, K, scale, act);
}

extern "C" void kernel_launch(void* const* d_in, const int* in_sizes, int n_in,
                              void* d_out, int out_size, void* d_ws, size_t ws_size,
                              hipStream_t stream)
{
    const float* images    = (const float*)d_in[0];
    const float* conv_w    = (const float*)d_in[1];
    const float* conv_b    = (const float*)d_in[2];
    const float* cls_token = (const float*)d_in[3];
    const float* wq = (const float*)d_in[4];  const float* bq = (const float*)d_in[5];
    const float* wk = (const float*)d_in[6];  const float* bk = (const float*)d_in[7];
    const float* wv = (const float*)d_in[8];  const float* bv = (const float*)d_in[9];
    const float* wo = (const float*)d_in[10]; const float* bo = (const float*)d_in[11];
    const float* ln1_s = (const float*)d_in[12]; const float* ln1_b = (const float*)d_in[13];
    const float* ln2_s = (const float*)d_in[14]; const float* ln2_b = (const float*)d_in[15];
    const float* w1 = (const float*)d_in[16]; const float* b1 = (const float*)d_in[17];
    const float* w2 = (const float*)d_in[18]; const float* b2 = (const float*)d_in[19];
    const float* wc = (const float*)d_in[20]; const float* bc = (const float*)d_in[21];
    float* out = (float*)d_out;

    char* ws = (char*)d_ws;
    const size_t PADH = (size_t)BM * 4096 * 2;   // padding (bytes)
    auto alloc = [&](size_t bytes) -> void* {
        void* p = (void*)ws;
        ws += (bytes + PADH + 255) & ~(size_t)255;
        return p;
    };
    const long long TOK = (long long)MTOK * Dd;             // 6304*768
    _Float16* wq16 = (_Float16*)alloc((size_t)Dd * Dd * 2);    // [N][K] transposed
    _Float16* wk16 = (_Float16*)alloc((size_t)Dd * Dd * 2);
    _Float16* wv16 = (_Float16*)alloc((size_t)Dd * Dd * 2);
    _Float16* wo16 = (_Float16*)alloc((size_t)Dd * Dd * 2);
    _Float16* w116 = (_Float16*)alloc((size_t)Dd * FFf * 2);   // [FF][D]
    _Float16* w216 = (_Float16*)alloc((size_t)FFf * Dd * 2);   // [D][FF]
    _Float16* cw16 = (_Float16*)alloc((size_t)Dd * Dd * 2);    // conv_w already [N=d][K]
    _Float16* wc16 = (_Float16*)alloc((size_t)Dd * NCc * 2);   // [NC][D]
    _Float16* Ap   = (_Float16*)alloc((size_t)Bz * NPATCH * Dd * 2);
    float*    xp   = (float*)   alloc((size_t)Bz * NPATCH * Dd * 4);
    float*    x    = (float*)   alloc((size_t)TOK * 4);
    _Float16* hbuf = (_Float16*)alloc((size_t)TOK * 2);
    _Float16* qf   = (_Float16*)alloc((size_t)TOK * 2);
    _Float16* kf   = (_Float16*)alloc((size_t)TOK * 2);
    _Float16* vf   = (_Float16*)alloc((size_t)TOK * 2);
    _Float16* Qh   = (_Float16*)alloc((size_t)TOK * 2);
    _Float16* Kh   = (_Float16*)alloc((size_t)TOK * 2);
    _Float16* VhT  = (_Float16*)alloc((size_t)Bz * Hh * HDd * SP * 2);
    float*    scor = (float*)   alloc((size_t)Bz * Hh * Ss * Ss * 4);
    _Float16* Pm   = (_Float16*)alloc((size_t)Bz * Hh * Ss * SP * 2);
    _Float16* Oh   = (_Float16*)alloc((size_t)TOK * 2);
    _Float16* Om   = (_Float16*)alloc((size_t)TOK * 2);
    _Float16* ff   = (_Float16*)alloc((size_t)MTOK * FFf * 2);
    _Float16* cls16= (_Float16*)alloc((size_t)Bz * Dd * 2);
    (void)ws_size; (void)in_sizes; (void)n_in; (void)out_size;

    const int TB = 256;
    const long long nTok = TOK;

    // ---- patch embedding as GEMM ----
    k_im2col<<<(unsigned)ceilDiv((long long)Bz * NPATCH * Dd, TB), TB, 0, stream>>>(images, Ap);
    k_f32_to_f16<<<(unsigned)ceilDiv((long long)Dd * Dd, TB), TB, 0, stream>>>(conv_w, cw16, (long long)Dd * Dd);
    k_transpose_to_f16<<<(unsigned)ceilDiv((long long)Dd * NCc, TB), TB, 0, stream>>>(wc, wc16, Dd, NCc);
    gemm(stream, Ap, Dd, 0, cw16, Dd, 0, xp, nullptr, Dd, 0, conv_b, nullptr,
         Bz * NPATCH, Dd, Dd, 1.0f, 0, 1);
    k_assemble_x<<<(unsigned)ceilDiv(nTok, TB), TB, 0, stream>>>(xp, cls_token, x);

    // ---- transformer layers ----
    for (int l = 0; l < Ll; ++l) {
        const long long oDD = (long long)l * Dd * Dd;
        const long long oDF = (long long)l * Dd * FFf;
        const unsigned gDD = (unsigned)ceilDiv((long long)Dd * Dd, TB);
        const unsigned gDF = (unsigned)ceilDiv((long long)Dd * FFf, TB);
        k_transpose_to_f16<<<gDD, TB, 0, stream>>>(wq + oDD, wq16, Dd, Dd);
        k_transpose_to_f16<<<gDD, TB, 0, stream>>>(wk + oDD, wk16, Dd, Dd);
        k_transpose_to_f16<<<gDD, TB, 0, stream>>>(wv + oDD, wv16, Dd, Dd);
        k_transpose_to_f16<<<gDD, TB, 0, stream>>>(wo + oDD, wo16, Dd, Dd);
        k_transpose_to_f16<<<gDF, TB, 0, stream>>>(w1 + oDF, w116, Dd, FFf);   // -> [FF][D]
        k_transpose_to_f16<<<gDF, TB, 0, stream>>>(w2 + oDF, w216, FFf, Dd);   // -> [D][FF]

        // LN1 -> h (f16)
        k_layernorm<<<MTOK, TB, 0, stream>>>(x, ln1_s + (long long)l * Dd, ln1_b + (long long)l * Dd, hbuf);

        // QKV projections (bias fused, f16 out)
        gemm(stream, hbuf, Dd, 0, wq16, Dd, 0, nullptr, qf, Dd, 0, bq + (long long)l * Dd, nullptr,
             MTOK, Dd, Dd, 1.0f, 0, 1);
        gemm(stream, hbuf, Dd, 0, wk16, Dd, 0, nullptr, kf, Dd, 0, bk + (long long)l * Dd, nullptr,
             MTOK, Dd, Dd, 1.0f, 0, 1);
        gemm(stream, hbuf, Dd, 0, wv16, Dd, 0, nullptr, vf, Dd, 0, bv + (long long)l * Dd, nullptr,
             MTOK, Dd, Dd, 1.0f, 0, 1);

        k_qkv_split<<<(unsigned)ceilDiv((long long)Bz * Hh * HDd * SP, TB), TB, 0, stream>>>(
            qf, kf, vf, Qh, Kh, VhT);

        // scores = Q @ K^T / sqrt(64)  (B^T of this GEMM is Kh itself)
        gemm(stream, Qh, HDd, (long long)Ss * HDd, Kh, HDd, (long long)Ss * HDd,
             scor, nullptr, Ss, (long long)Ss * Ss, nullptr, nullptr,
             Ss, Ss, HDd, 0.125f, 0, Bz * Hh);

        k_softmax<<<Bz * Hh * Ss, TB, 0, stream>>>(scor, Pm);

        // O = P @ V  (K = SP = 224, zero padded in both Pm cols and VhT cols)
        gemm(stream, Pm, SP, (long long)Ss * SP, VhT, SP, (long long)HDd * SP,
             nullptr, Oh, HDd, (long long)Ss * HDd, nullptr, nullptr,
             Ss, HDd, SP, 1.0f, 0, Bz * Hh);

        k_merge_heads<<<(unsigned)ceilDiv(nTok, TB), TB, 0, stream>>>(Oh, Om);

        // x = x + O @ wo + bo  (residual fused)
        gemm(stream, Om, Dd, 0, wo16, Dd, 0, x, nullptr, Dd, 0, bo + (long long)l * Dd, x,
             MTOK, Dd, Dd, 1.0f, 0, 1);

        // LN2 -> h (f16)
        k_layernorm<<<MTOK, TB, 0, stream>>>(x, ln2_s + (long long)l * Dd, ln2_b + (long long)l * Dd, hbuf);

        // ff = gelu(h @ w1 + b1)  (f16 out, exact GELU)
        gemm(stream, hbuf, Dd, 0, w116, Dd, 0, nullptr, ff, FFf, 0, b1 + (long long)l * FFf, nullptr,
             MTOK, FFf, Dd, 1.0f, 1, 1);

        // x = x + ff @ w2 + b2  (residual fused)
        gemm(stream, ff, FFf, 0, w216, FFf, 0, x, nullptr, Dd, 0, b2 + (long long)l * Dd, x,
             MTOK, Dd, FFf, 1.0f, 0, 1);
    }

    // ---- classifier head on cls token ----
    k_extract_cls<<<(Bz * Dd + TB - 1) / TB, TB, 0, stream>>>(x, cls16);
    gemm(stream, cls16, Dd, 0, wc16, Dd, 0, out, nullptr, NCc, 0, bc, nullptr,
         Bz, NCc, Dd, 1.0f, 0, 1);
}
